// ResBlock_d3d_47863115546760
// MI455X (gfx1250) — matrix operand order repaired and checked by reading, not compiled, not run on hardware
//
#include <hip/hip_runtime.h>
#include <stdint.h>

// ---------------------------------------------------------------------------
// ResBlock with two deformable 3D convs (offsets in H,W only) for MI455X.
// Strategy: everything is L2-resident (~15MB); the 4 big contractions are
// K=1728 GEMMs done with v_wmma_f32_16x16x32_bf16 (f32 accumulate).
// Activations are kept channels-last bf16 so every bilinear corner gather is
// a contiguous 16B load covering 8 channels (one address calc per tap/pixel
// amortized over 64 channels). TDM is not applicable: gather addresses are
// data-dependent per pixel (predicted offsets).
// ---------------------------------------------------------------------------

typedef __attribute__((ext_vector_type(16))) __bf16 v16bf;
typedef __attribute__((ext_vector_type(8)))  float  v8f;

#define T_   7
#define H_   64
#define W_   64
#define C_   64
#define THW  (T_ * H_ * W_)          // 28672 spatial points
#define KK_  27                      // 3x3x3 taps
#define KCH  54                      // K = 27*64 = 1728 -> 54 chunks of 32
#define MT   4                       // 4 M-tiles of 16 rows (64 rows, O padded)
#define APACK_ELEMS (MT * KCH * 32 * 16)   // 110592 bf16 per weight tensor

// ---- bf16 helpers (manual, storage = unsigned short) ----------------------
__device__ __forceinline__ unsigned bfround(float x) {   // f32 -> bf16 (RNE)
    unsigned u = __float_as_uint(x);
    u += 0x7FFFu + ((u >> 16) & 1u);
    return u >> 16;
}
__device__ __forceinline__ unsigned packbf2(float lo, float hi) {
    return bfround(lo) | (bfround(hi) << 16);
}
__device__ __forceinline__ float bflo(unsigned u) { return __uint_as_float(u << 16); }
__device__ __forceinline__ float bfhi(unsigned u) { return __uint_as_float(u & 0xFFFF0000u); }
__device__ __forceinline__ int imin(int a, int b) { return a < b ? a : b; }
__device__ __forceinline__ int imax(int a, int b) { return a > b ? a : b; }

union FragBF {            // one WMMA 16-bit operand fragment per lane (32B)
    v16bf v;
    uint4 q[2];
    unsigned u[8];
};

// ---------------------------------------------------------------------------
// x (C,T,H,W) f32  ->  xcl (T,H,W,C) bf16   (writes coalesced)
// ---------------------------------------------------------------------------
__global__ void k_to_cl(const float* __restrict__ x,
                        unsigned short* __restrict__ xcl, int n) {
    int i = blockIdx.x * blockDim.x + threadIdx.x;
    if (i >= n) return;
    int c = i & (C_ - 1);
    int s = i >> 6;
    xcl[i] = (unsigned short)bfround(x[(size_t)c * THW + s]);
}

// ---------------------------------------------------------------------------
// Pack OIDHW f32 weights into the A-fragment memory image.
// A 16x32 layout (matches B): row M = lane&15; lane-half 0 holds
// K in {0..15}, lane-half 1 holds K in {16..31}.
// Global K index = tap*64 + c  (matches weight.reshape(O, C, KK) einsum).
// ---------------------------------------------------------------------------
__global__ void k_pack_w(const float* __restrict__ w,
                         unsigned short* __restrict__ apack, int ovalid) {
    int i = blockIdx.x * blockDim.x + threadIdx.x;
    if (i >= APACK_ELEMS) return;
    int e     = i & 15;
    int lane  = (i >> 4) & 31;
    int rest  = i >> 9;
    int kc    = rest % KCH;
    int mt    = rest / KCH;
    int row   = lane & 15;
    int khalf = lane >> 4;
    int k     = e + khalf * 16;   // A-operand K permutation
    int K     = kc * 32 + k;
    int tap   = K >> 6;
    int c     = K & 63;
    int O     = mt * 16 + row;
    float val = (O < ovalid) ? w[((size_t)O * C_ + c) * KK_ + tap] : 0.f;
    apack[i] = (unsigned short)bfround(val);
}

// ---------------------------------------------------------------------------
// Implicit-GEMM 3x3x3 conv (offset prediction): out(s, O) for O<54.
// Block = 128 thr = 4 waves; wave mt owns rows [16mt,16mt+16), block owns 16
// consecutive spatial points (one W-run, W=64 divisible by 16).
// Per K-chunk of 32 the tap is constant, so the per-lane B fragment is one
// contiguous run of 16 channels (this code's B order: K = 16*lanehalf + e).
// ---------------------------------------------------------------------------
__global__ __launch_bounds__(128) void k_conv_off(
    const unsigned short* __restrict__ xcl,
    const unsigned short* __restrict__ apack,
    const float* __restrict__ bias,
    float* __restrict__ offcl) {
    const int lane  = threadIdx.x & 31;
    const int mt    = threadIdx.x >> 5;
    const int n     = lane & 15;
    const int khalf = lane >> 4;
    const int s     = blockIdx.x * 16 + n;
    const int w     = s & 63, h = (s >> 6) & 63, t = s >> 12;

    v8f acc = {};
    const unsigned short* ap = apack + ((size_t)mt * KCH * 32 + lane) * 16;

    for (int kc = 0; kc < KCH; ++kc) {
        const int tap = kc >> 1;
        const int kt = tap / 9 - 1, kh = (tap / 3) % 3 - 1, kw = tap % 3 - 1;
        const int tt = t + kt, hh = h + kh, ww = w + kw;
        const bool ok = (unsigned)tt < (unsigned)T_ && (unsigned)hh < (unsigned)H_ &&
                        (unsigned)ww < (unsigned)W_;
        FragBF a, b;
        a.q[0] = *(const uint4*)(ap + (size_t)kc * 512);
        a.q[1] = *(const uint4*)(ap + (size_t)kc * 512 + 8);
        if (ok) {
            const int c0 = (kc & 1) * 32 + khalf * 16;   // B: K = 16*khalf + e
            const unsigned short* p =
                xcl + (((size_t)(tt * H_ + hh) * W_ + ww) * C_ + c0);
            b.q[0] = *(const uint4*)p;
            b.q[1] = *(const uint4*)(p + 8);
        } else {
            uint4 z; z.x = z.y = z.z = z.w = 0u;
            b.q[0] = z; b.q[1] = z;
        }
        acc = __builtin_amdgcn_wmma_f32_16x16x32_bf16(
            false, a.v, false, b.v, (short)0, acc, false, false);
    }

    // D layout: column N = lane&15, row M = 8*khalf + r (+16*mt), r consecutive
    const int Ob = mt * 16 + khalf * 8;
    float* o = offcl + (size_t)s * 64 + Ob;
#pragma unroll
    for (int r = 0; r < 8; ++r) {
        const int O = Ob + r;
        o[r] = acc[r] + ((O < 2 * KK_) ? bias[O] : 0.f);
    }
}

// ---------------------------------------------------------------------------
// Deformable conv: per tap compute bilinear geometry once per (lane,pixel),
// gather 4 corner channel-runs, blend in f32, repack bf16, 2 WMMAs per tap.
// MODE 0: leaky-ReLU(0.1) -> ycl (channels-last bf16)
// MODE 1: + residual x    -> out (NCTHW f32)
// ---------------------------------------------------------------------------
template <int MODE>
__global__ __launch_bounds__(128) void k_deform(
    const unsigned short* __restrict__ xcl,
    const float* __restrict__ offcl,
    const unsigned short* __restrict__ apack,
    unsigned short* __restrict__ ycl,
    const float* __restrict__ xres,
    float* __restrict__ out) {
    const int lane  = threadIdx.x & 31;
    const int mt    = threadIdx.x >> 5;
    const int n     = lane & 15;
    const int khalf = lane >> 4;
    const int s     = blockIdx.x * 16 + n;
    const int w     = s & 63, h = (s >> 6) & 63, t = s >> 12;

    v8f acc = {};
    const unsigned short* ap   = apack + ((size_t)mt * KCH * 32 + lane) * 16;
    const float*          offp = offcl + (size_t)s * 64;

    for (int tap = 0; tap < KK_; ++tap) {
        const int kt = tap / 9 - 1, kh = (tap / 3) % 3 - 1, kw = tap % 3 - 1;
        const int tt  = t + kt;
        const bool tok = (unsigned)tt < (unsigned)T_;
        const int ttc = imin(imax(tt, 0), T_ - 1);

        const float offh = offp[2 * tap];
        const float offw = offp[2 * tap + 1];
        const float ph = (float)(h + kh) + offh;
        const float pw = (float)(w + kw) + offw;
        const float h0f = floorf(ph), w0f = floorf(pw);
        const float lh = ph - h0f, lw = pw - w0f;
        const int h0 = (int)h0f, w0i = (int)w0f;
        const int h1 = h0 + 1,   w1i = w0i + 1;

        float w00 = (1.f - lh) * (1.f - lw), w01 = (1.f - lh) * lw;
        float w10 = lh * (1.f - lw),         w11 = lh * lw;
        const bool h0ok = (unsigned)h0  < (unsigned)H_, h1ok = (unsigned)h1  < (unsigned)H_;
        const bool w0ok = (unsigned)w0i < (unsigned)W_, w1ok = (unsigned)w1i < (unsigned)W_;
        if (!(tok && h0ok && w0ok)) w00 = 0.f;
        if (!(tok && h0ok && w1ok)) w01 = 0.f;
        if (!(tok && h1ok && w0ok)) w10 = 0.f;
        if (!(tok && h1ok && w1ok)) w11 = 0.f;
        const int h0c = imin(imax(h0, 0), H_ - 1), h1c = imin(imax(h1, 0), H_ - 1);
        const int w0c = imin(imax(w0i, 0), W_ - 1), w1c = imin(imax(w1i, 0), W_ - 1);

        const size_t rb  = (size_t)ttc * (H_ * W_ * C_);
        const size_t b00 = rb + ((size_t)h0c * W_ + w0c) * C_;
        const size_t b01 = rb + ((size_t)h0c * W_ + w1c) * C_;
        const size_t b10 = rb + ((size_t)h1c * W_ + w0c) * C_;
        const size_t b11 = rb + ((size_t)h1c * W_ + w1c) * C_;

#pragma unroll
        for (int half = 0; half < 2; ++half) {
            const int kc    = tap * 2 + half;
            const int cbase = half * 32 + khalf * 16;   // B: K = 16*khalf + e
            FragBF a, b;
            a.q[0] = *(const uint4*)(ap + (size_t)kc * 512);
            a.q[1] = *(const uint4*)(ap + (size_t)kc * 512 + 8);
#pragma unroll
            for (int g = 0; g < 2; ++g) {               // elements 8g..8g+7
                const int c0 = cbase + g * 8;
                const uint4 q00 = *(const uint4*)(xcl + b00 + c0);
                const uint4 q01 = *(const uint4*)(xcl + b01 + c0);
                const uint4 q10 = *(const uint4*)(xcl + b10 + c0);
                const uint4 q11 = *(const uint4*)(xcl + b11 + c0);
                const unsigned a00[4] = {q00.x, q00.y, q00.z, q00.w};
                const unsigned a01[4] = {q01.x, q01.y, q01.z, q01.w};
                const unsigned a10[4] = {q10.x, q10.y, q10.z, q10.w};
                const unsigned a11[4] = {q11.x, q11.y, q11.z, q11.w};
#pragma unroll
                for (int j = 0; j < 4; ++j) {
                    const float lo = w00 * bflo(a00[j]) + w01 * bflo(a01[j]) +
                                     w10 * bflo(a10[j]) + w11 * bflo(a11[j]);
                    const float hi = w00 * bfhi(a00[j]) + w01 * bfhi(a01[j]) +
                                     w10 * bfhi(a10[j]) + w11 * bfhi(a11[j]);
                    b.u[g * 4 + j] = packbf2(lo, hi);
                }
            }
            acc = __builtin_amdgcn_wmma_f32_16x16x32_bf16(
                false, a.v, false, b.v, (short)0, acc, false, false);
        }
    }

    const int Ob = mt * 16 + khalf * 8;
    if (MODE == 0) {
        unsigned short* yp = ycl + (size_t)s * 64 + Ob;
#pragma unroll
        for (int r = 0; r < 8; ++r) {
            float v = acc[r];
            v = (v > 0.f) ? v : 0.1f * v;               // leaky_relu(0.1)
            yp[r] = (unsigned short)bfround(v);
        }
    } else {
#pragma unroll
        for (int r = 0; r < 8; ++r) {
            const int O = Ob + r;
            out[(size_t)O * THW + s] = acc[r] + xres[(size_t)O * THW + s];
        }
    }
}

// ---------------------------------------------------------------------------
extern "C" void kernel_launch(void* const* d_in, const int* in_sizes, int n_in,
                              void* d_out, int out_size, void* d_ws, size_t ws_size,
                              hipStream_t stream) {
    (void)in_sizes; (void)n_in; (void)out_size; (void)ws_size;
    const float* x      = (const float*)d_in[0];
    const float* w_off0 = (const float*)d_in[1];
    const float* b_off0 = (const float*)d_in[2];
    const float* w0     = (const float*)d_in[3];
    const float* w_off1 = (const float*)d_in[4];
    const float* b_off1 = (const float*)d_in[5];
    const float* w1     = (const float*)d_in[6];
    float* out = (float*)d_out;

    char* p = (char*)d_ws;
    auto carve = [&](size_t bytes) -> char* {
        char* r = p;
        p += (bytes + 255) & ~(size_t)255;
        return r;
    };
    unsigned short* xcl    = (unsigned short*)carve((size_t)THW * C_ * 2);
    unsigned short* ycl    = (unsigned short*)carve((size_t)THW * C_ * 2);
    float*          offcl  = (float*)carve((size_t)THW * 64 * 4);   // reused
    unsigned short* apOff0 = (unsigned short*)carve((size_t)APACK_ELEMS * 2);
    unsigned short* apW0   = (unsigned short*)carve((size_t)APACK_ELEMS * 2);
    unsigned short* apOff1 = (unsigned short*)carve((size_t)APACK_ELEMS * 2);
    unsigned short* apW1   = (unsigned short*)carve((size_t)APACK_ELEMS * 2);

    const int NE = THW * C_;                       // 1,835,008
    k_to_cl<<<(NE + 255) / 256, 256, 0, stream>>>(x, xcl, NE);

    const int NPB = (APACK_ELEMS + 255) / 256;
    k_pack_w<<<NPB, 256, 0, stream>>>(w_off0, apOff0, 2 * KK_);
    k_pack_w<<<NPB, 256, 0, stream>>>(w0,     apW0,   C_);
    k_pack_w<<<NPB, 256, 0, stream>>>(w_off1, apOff1, 2 * KK_);
    k_pack_w<<<NPB, 256, 0, stream>>>(w1,     apW1,   C_);

    const int NT = THW / 16;                       // 1792 tiles
    k_conv_off<<<NT, 128, 0, stream>>>(xcl, apOff0, b_off0, offcl);
    k_deform<0><<<NT, 128, 0, stream>>>(xcl, offcl, apW0, ycl, nullptr, nullptr);
    k_conv_off<<<NT, 128, 0, stream>>>(ycl, apOff1, b_off1, offcl);
    k_deform<1><<<NT, 128, 0, stream>>>(ycl, offcl, apW1, nullptr, x, out);
}
